// TalkieFullSequenceOnnxWrapper_53661321396790
// MI455X (gfx1250) — matrix operand order repaired and checked by reading, not compiled, not run on hardware
//
#include <hip/hip_runtime.h>
#include <hip/hip_bf16.h>
#include <stdint.h>

// ---------------------------------------------------------------------------
// MI455X (gfx1250) transformer forward.
// Strategy: fp32 weights are streamed from HBM exactly once (≈390 MB ≈ 17 µs
// @ 23.3 TB/s), converted to bf16 while staging into LDS, and all matmuls run
// on v_wmma_f32_16x16x32_bf16 (the high-throughput CDNA5 matrix path).
// bf16 activation tiles are staged into LDS with the CDNA5 async LDS-DMA
// (global_load_async_to_lds_b128, ASYNCcnt), overlapping the fp32->bf16
// weight conversion that must go through VGPRs.
// Attention is a flash-style single-wave kernel (online softmax + WMMA).
// ---------------------------------------------------------------------------

#define SEQ   1024
#define DIM   1024
#define NH    16
#define HD    64
#define DFF   4096
#define NL    4
#define VOC   32000

typedef __bf16 v16bf __attribute__((ext_vector_type(16)));
typedef float  v8f   __attribute__((ext_vector_type(8)));

union FragU { uint4 q[2]; v16bf v; };

static __device__ __forceinline__ unsigned short f2bf(float f) {
    union { float f; unsigned u; } v; v.f = f;
    unsigned u = v.u;
    unsigned r = u + 0x7FFFu + ((u >> 16) & 1u);   // round-to-nearest-even
    return (unsigned short)(r >> 16);
}

static __device__ __forceinline__ uint4 pack8(float4 a, float4 b) {
    uint4 r;
    r.x = (unsigned)f2bf(a.x) | ((unsigned)f2bf(a.y) << 16);
    r.y = (unsigned)f2bf(a.z) | ((unsigned)f2bf(a.w) << 16);
    r.z = (unsigned)f2bf(b.x) | ((unsigned)f2bf(b.y) << 16);
    r.w = (unsigned)f2bf(b.z) | ((unsigned)f2bf(b.w) << 16);
    return r;
}

static __device__ __forceinline__ v8f wmma_bf16(v16bf a, v16bf b, v8f c) {
    return __builtin_amdgcn_wmma_f32_16x16x32_bf16(
        /*neg_a=*/false, a, /*neg_b=*/false, b,
        /*c_mod=*/(short)0, c, /*reuse_a=*/false, /*reuse_b=*/false);
}

// Async global->LDS 16B copy (CDNA5 LDS-DMA, tracked by ASYNCcnt).
// ldsOff: LDS byte address (low 32 bits of the generic shared pointer),
// gaddr : 64-bit global address. Per-lane addressing.
static __device__ __forceinline__ void async_ld16(unsigned ldsOff, const void* gaddr) {
#if defined(__gfx1250__)
    asm volatile("global_load_async_to_lds_b128 %0, %1, off"
                 :: "v"(ldsOff), "v"((unsigned long long)(uintptr_t)gaddr)
                 : "memory");
#else
    *(uint4*)(uintptr_t)ldsOff = *(const uint4*)gaddr;   // never used on gfx1250
#endif
}

static __device__ __forceinline__ void wait_asynccnt0() {
#if defined(__gfx1250__)
    asm volatile("s_wait_asynccnt 0x0" ::: "memory");
#endif
}

// ---------------------------------------------------------------------------
// GEMM: C[M,N] = alpha * A[M,K](bf16) * W[N,K](fp32, converted to bf16)^T
// Tile 128x128x32, 256 threads (8 waves), wave tile 32x64 -> 8 wmma / K-step.
// A tile:   async LDS-DMA (no VGPR round trip, overlaps with W conversion).
// W tile:   fp32 global load -> bf16 pack -> ds_store.
// Requires M%128==0, N%128==0, K%32==0 (true for all uses here).
// ---------------------------------------------------------------------------
__global__ __launch_bounds__(256) void k_gemm(
    const unsigned short* __restrict__ A, const float* __restrict__ W,
    float* __restrict__ C, int M, int N, int K, const float* alphaPtr)
{
    __shared__ unsigned short sA[128 * 40];   // 40-elem (80B) padded rows
    __shared__ unsigned short sB[128 * 40];

    const int tid    = threadIdx.x;
    const int lane   = tid & 31;
    const int wave   = tid >> 5;
    const int wm     = (wave & 3) * 32;       // wave M offset in tile
    const int wn     = (wave >> 2) * 64;      // wave N offset in tile
    const int lane15 = lane & 15;
    const int laneHi = (lane >> 4) & 1;
    const int tileN  = blockIdx.x * 128;
    const int tileM  = blockIdx.y * 128;
    const int row2   = tid >> 1;              // 0..127 : row handled for staging
    const int seg    = tid & 1;               // 16-element segment within row

    v8f acc[2][4];
#pragma unroll
    for (int i = 0; i < 2; ++i)
#pragma unroll
        for (int j = 0; j < 4; ++j)
#pragma unroll
            for (int v = 0; v < 8; ++v) acc[i][j][v] = 0.0f;

    const unsigned short* Ag = A + (size_t)(tileM + row2) * K + seg * 16;
    const float*          Bg = W + (size_t)(tileN + row2) * K + seg * 16;
    const int sOff = row2 * 40 + seg * 16;
    const unsigned ldsA0 = (unsigned)(uintptr_t)&sA[sOff];
    const unsigned ldsA1 = (unsigned)(uintptr_t)&sA[sOff + 8];

    for (int k0 = 0; k0 < K; k0 += 32) {
        // ---- A tile: async LDS-DMA (bf16 passthrough, ASYNCcnt) ----
        async_ld16(ldsA0, Ag + k0);
        async_ld16(ldsA1, Ag + k0 + 8);

        // ---- W tile: fp32 -> bf16 through VGPRs ----
        float4 w0 = *(const float4*)(Bg + k0);
        float4 w1 = *(const float4*)(Bg + k0 + 4);
        float4 w2 = *(const float4*)(Bg + k0 + 8);
        float4 w3 = *(const float4*)(Bg + k0 + 12);
        if (k0 + 32 < K) {
            __builtin_prefetch(Bg + k0 + 32, 0, 1);   // global_prefetch_b8
            __builtin_prefetch(Ag + k0 + 32, 0, 1);
        }
        *(uint4*)&sB[sOff]     = pack8(w0, w1);
        *(uint4*)&sB[sOff + 8] = pack8(w2, w3);

        wait_asynccnt0();         // drain LDS-DMA before the barrier
        __syncthreads();

        // ---- fragments (ISA 7.12.2 layouts) ----
        v16bf aF[2], bF[4];
#pragma unroll
        for (int i = 0; i < 2; ++i) {
            // A 16x32: lane<16 holds K 0-7 & 16-23, lane>=16 holds K 8-15 & 24-31
            int base = (wm + i * 16 + lane15) * 40 + laneHi * 8;
            FragU u;
            u.q[0] = *(const uint4*)&sA[base];
            u.q[1] = *(const uint4*)&sA[base + 16];
            aF[i] = u.v;
        }
#pragma unroll
        for (int j = 0; j < 4; ++j) {
            // B 32x16: lane = column, lane<16 holds K 0-7 & 16-23, lane>=16 holds K 8-15 & 24-31
            int base = (wn + j * 16 + lane15) * 40 + laneHi * 8;
            FragU u;
            u.q[0] = *(const uint4*)&sB[base];
            u.q[1] = *(const uint4*)&sB[base + 16];
            bF[j] = u.v;
        }
#pragma unroll
        for (int i = 0; i < 2; ++i)
#pragma unroll
            for (int j = 0; j < 4; ++j)
                acc[i][j] = wmma_bf16(aF[i], bF[j], acc[i][j]);
        __syncthreads();
    }

    const float alpha = alphaPtr ? *alphaPtr : 1.0f;
#pragma unroll
    for (int i = 0; i < 2; ++i)
#pragma unroll
        for (int j = 0; j < 4; ++j) {
            int n = tileN + wn + j * 16 + lane15;
#pragma unroll
            for (int v = 0; v < 8; ++v) {
                int m = tileM + wm + i * 16 + v + laneHi * 8;
                C[(size_t)m * N + n] = acc[i][j][v] * alpha;
            }
        }
}

// ---------------------------------------------------------------------------
// Embedding gather + RMS norm -> x and e_x (fp32). One block per token.
// ---------------------------------------------------------------------------
__global__ __launch_bounds__(256) void k_embed_rms(
    const int* __restrict__ ids, const float* __restrict__ embed,
    float* __restrict__ x, float* __restrict__ ex)
{
    const int s = blockIdx.x, t = threadIdx.x;
    const float* e = embed + (size_t)ids[s] * DIM;
    float vals[4], ss = 0.f;
#pragma unroll
    for (int i = 0; i < 4; ++i) { vals[i] = e[t + i * 256]; ss += vals[i] * vals[i]; }
    __shared__ float red[256];
    red[t] = ss; __syncthreads();
    for (int w = 128; w > 0; w >>= 1) { if (t < w) red[t] += red[t + w]; __syncthreads(); }
    const float sc = rsqrtf(red[0] * (1.0f / DIM) + 1e-5f);
#pragma unroll
    for (int i = 0; i < 4; ++i) {
        float o = vals[i] * sc;
        x[(size_t)s * DIM + t + i * 256]  = o;
        ex[(size_t)s * DIM + t + i * 256] = o;
    }
}

// RMS norm of a fp32 (SEQ x DIM) row -> bf16 row. One block per token.
__global__ __launch_bounds__(256) void k_rms_bf16(
    const float* __restrict__ in, unsigned short* __restrict__ out)
{
    const int s = blockIdx.x, t = threadIdx.x;
    const float* r = in + (size_t)s * DIM;
    float vals[4], ss = 0.f;
#pragma unroll
    for (int i = 0; i < 4; ++i) { vals[i] = r[t + i * 256]; ss += vals[i] * vals[i]; }
    __shared__ float red[256];
    red[t] = ss; __syncthreads();
    for (int w = 128; w > 0; w >>= 1) { if (t < w) red[t] += red[t + w]; __syncthreads(); }
    const float sc = rsqrtf(red[0] * (1.0f / DIM) + 1e-5f);
#pragma unroll
    for (int i = 0; i < 4; ++i)
        out[(size_t)s * DIM + t + i * 256] = f2bf(vals[i] * sc);
}

// ---------------------------------------------------------------------------
// Q/K/V post-processing. mode 0: q (RoPE + head-RMS + head_gain) -> (h,s,d)
//                        mode 1: k (RoPE + head-RMS)             -> (h,s,d)
//                        mode 2: v (plain bf16)                  -> (h,d,s)  (transposed!)
// grid (SEQ, NH/4), block (64, 4)
// ---------------------------------------------------------------------------
__global__ __launch_bounds__(256) void k_qkv_post(
    const float* __restrict__ src, unsigned short* __restrict__ dst,
    const float* __restrict__ gain, int mode)
{
    const int s = blockIdx.x;
    const int h = blockIdx.y * 4 + threadIdx.y;
    const int d = threadIdx.x;
    const size_t base = (size_t)s * DIM + h * HD;

    if (mode == 2) {
        dst[((size_t)h * HD + d) * SEQ + s] = f2bf(src[base + d]);
        return;
    }
    // RoPE
    const int i = d & 31;
    const float fr  = __expf(-(float)(2 * i) * (9.210340371976184f / 64.0f)); // theta^(-2i/64)
    const float ang = (float)s * fr;
    const float c = __cosf(ang), sn = __sinf(ang);
    const float x1 = src[base + i];
    const float x2 = src[base + i + 32];
    const float r = (d < 32) ? (x1 * c + x2 * sn) : (-x1 * sn + x2 * c);

    __shared__ float buf[4][64];
    buf[threadIdx.y][d] = r * r;
    __syncthreads();
    float ss = 0.f;
#pragma unroll
    for (int k = 0; k < 64; ++k) ss += buf[threadIdx.y][k];
    float o = r * rsqrtf(ss * (1.0f / HD) + 1e-5f);
    if (mode == 0) o *= gain[h];
    dst[((size_t)h * SEQ + s) * HD + d] = f2bf(o);
}

// ---------------------------------------------------------------------------
// Flash attention: one wave per (16-row Q tile, head). Online softmax.
// qh/kh: (h, s, d) bf16 ; vT: (h, d, s) bf16 ; y: (s, D) fp32.
// ---------------------------------------------------------------------------
__global__ __launch_bounds__(32) void k_attn(
    const unsigned short* __restrict__ qh, const unsigned short* __restrict__ kh,
    const unsigned short* __restrict__ vT, float* __restrict__ y)
{
    __shared__ unsigned short sP[16 * 40];

    const int lane   = threadIdx.x;
    const int lane15 = lane & 15;
    const int laneHi = (lane >> 4) & 1;
    const int qbase  = blockIdx.x * 16;
    const int h      = blockIdx.y;

    // Q fragments (16 x 64 split in two 16x32 A-fragments), straight from global
    v16bf aQ[2];
    const unsigned short* qrow = qh + ((size_t)h * SEQ + qbase) * HD;
#pragma unroll
    for (int t = 0; t < 2; ++t) {
        const int kc = t * 32 + laneHi * 8;
        FragU u;
        u.q[0] = *(const uint4*)(qrow + lane15 * HD + kc);
        u.q[1] = *(const uint4*)(qrow + lane15 * HD + kc + 16);
        aQ[t] = u.v;
    }

    float rowM[8], rowL[8];
    v8f O[4];
#pragma unroll
    for (int v = 0; v < 8; ++v) { rowM[v] = -1e30f; rowL[v] = 0.0f; }
#pragma unroll
    for (int j = 0; j < 4; ++j)
#pragma unroll
        for (int v = 0; v < 8; ++v) O[j][v] = 0.0f;

    const int nblk = (qbase + 16 + 31) >> 5;     // 32-key blocks covering causal span
    for (int blk = 0; blk < nblk; ++blk) {
        const int kb0 = blk * 32;

        // ---- scores = Q * K^T (16 x 32) ----
        v8f sf[2];
#pragma unroll
        for (int sub = 0; sub < 2; ++sub)
#pragma unroll
            for (int v = 0; v < 8; ++v) sf[sub][v] = 0.0f;
#pragma unroll
        for (int t = 0; t < 2; ++t)
#pragma unroll
            for (int sub = 0; sub < 2; ++sub) {
                const unsigned short* kr =
                    kh + ((size_t)h * SEQ + kb0 + sub * 16 + lane15) * HD + t * 32 + laneHi * 8;
                FragU u;
                u.q[0] = ((const uint4*)kr)[0];
                u.q[1] = ((const uint4*)kr)[2];
                sf[sub] = wmma_bf16(aQ[t], u.v, sf[sub]);
            }

        // ---- online softmax ----
        float pr[2][8];
#pragma unroll
        for (int v = 0; v < 8; ++v) {
            const int mq = qbase + v + laneHi * 8;
            float s0 = sf[0][v] * 0.125f;
            float s1 = sf[1][v] * 0.125f;
            if (kb0 + lane15 > mq)      s0 = -1e30f;
            if (kb0 + 16 + lane15 > mq) s1 = -1e30f;
            float mx = fmaxf(s0, s1);
#pragma unroll
            for (int off = 1; off < 16; off <<= 1)
                mx = fmaxf(mx, __shfl_xor(mx, off, 32));
            const float nm = fmaxf(rowM[v], mx);
            const float al = __expf(rowM[v] - nm);
            const float p0 = __expf(s0 - nm);
            const float p1 = __expf(s1 - nm);
            float rs = p0 + p1;
#pragma unroll
            for (int off = 1; off < 16; off <<= 1)
                rs += __shfl_xor(rs, off, 32);
            rowL[v] = rowL[v] * al + rs;
            rowM[v] = nm;
            pr[0][v] = p0; pr[1][v] = p1;
#pragma unroll
            for (int j = 0; j < 4; ++j) O[j][v] = O[j][v] * al;
        }

        // ---- stage P (bf16) into LDS in C-layout, reload as A-fragment ----
#pragma unroll
        for (int sub = 0; sub < 2; ++sub)
#pragma unroll
            for (int v = 0; v < 8; ++v)
                sP[(v + laneHi * 8) * 40 + sub * 16 + lane15] = f2bf(pr[sub][v]);
        __syncthreads();
        v16bf aP;
        {
            const int kc = laneHi * 8;
            FragU u;
            u.q[0] = *(const uint4*)&sP[lane15 * 40 + kc];
            u.q[1] = *(const uint4*)&sP[lane15 * 40 + kc + 16];
            aP = u.v;
        }

        // ---- O += P * V (V stored transposed -> contiguous B fragments) ----
#pragma unroll
        for (int j = 0; j < 4; ++j) {
            const unsigned short* vr =
                vT + ((size_t)h * HD + j * 16 + lane15) * SEQ + kb0 + laneHi * 8;
            FragU u;
            u.q[0] = ((const uint4*)vr)[0];
            u.q[1] = ((const uint4*)vr)[2];
            O[j] = wmma_bf16(aP, u.v, O[j]);
        }
        __syncthreads();
    }

    // ---- write y = O / l ----
#pragma unroll
    for (int j = 0; j < 4; ++j) {
        const int n = h * HD + j * 16 + lane15;
#pragma unroll
        for (int v = 0; v < 8; ++v) {
            const int m = qbase + v + laneHi * 8;
            y[(size_t)m * DIM + n] = O[j][v] / rowL[v];
        }
    }
}

// ---------------------------------------------------------------------------
// small elementwise kernels
// ---------------------------------------------------------------------------
__global__ void k_to_bf16(const float* __restrict__ in, unsigned short* __restrict__ out, int n) {
    int i = blockIdx.x * blockDim.x + threadIdx.x;
    if (i < n) out[i] = f2bf(in[i]);
}

__global__ void k_add_scaled(float* __restrict__ x, const float* __restrict__ c,
                             const float* __restrict__ g, int n) {
    int i = blockIdx.x * blockDim.x + threadIdx.x;
    if (i < n) x[i] += c[i] * (*g);
}

__global__ void k_silu_mul(const float* __restrict__ g, const float* __restrict__ l,
                           unsigned short* __restrict__ out, int n) {
    int i = blockIdx.x * blockDim.x + threadIdx.x;
    if (i < n) {
        float gv = g[i];
        float s  = gv / (1.0f + __expf(-gv));
        out[i] = f2bf(s * l[i]);
    }
}

// ---------------------------------------------------------------------------
// driver
// ---------------------------------------------------------------------------
extern "C" void kernel_launch(void* const* d_in, const int* in_sizes, int n_in,
                              void* d_out, int out_size, void* d_ws, size_t ws_size,
                              hipStream_t stream) {
    (void)in_sizes; (void)n_in; (void)out_size; (void)ws_size;

    const int*   ids      = (const int*)d_in[0];
    const float* embed    = (const float*)d_in[1];
    const float* Wq       = (const float*)d_in[2];
    const float* Wk       = (const float*)d_in[3];
    const float* Wv       = (const float*)d_in[4];
    const float* Wo       = (const float*)d_in[5];
    const float* head_g   = (const float*)d_in[6];
    const float* attn_g   = (const float*)d_in[7];
    const float* Wg       = (const float*)d_in[8];
    const float* Wl       = (const float*)d_in[9];
    const float* Wr       = (const float*)d_in[10];
    const float* mlp_g    = (const float*)d_in[11];
    const float* emb_skip = (const float*)d_in[12];
    const float* lm_head  = (const float*)d_in[13];
    const float* lm_gain  = (const float*)d_in[14];

    char* ws = (char*)d_ws;
    const size_t MB = 1u << 20;
    float*          x    = (float*)         (ws + 0 * MB);   // SEQ x DIM  (4 MB)
    float*          ex   = (float*)         (ws + 4 * MB);   // SEQ x DIM  (4 MB)
    unsigned short* abf  = (unsigned short*)(ws + 8 * MB);   // SEQ x DIM bf16 (2 MB)
    float*          qf   = (float*)         (ws + 10 * MB);  // SEQ x DIM
    float*          kf   = (float*)         (ws + 14 * MB);
    float*          vf   = (float*)         (ws + 18 * MB);
    unsigned short* qhB  = (unsigned short*)(ws + 22 * MB);  // (h,s,d) bf16
    unsigned short* khB  = (unsigned short*)(ws + 24 * MB);
    unsigned short* vTB  = (unsigned short*)(ws + 26 * MB);  // (h,d,s) bf16
    float*          yF   = (float*)         (ws + 28 * MB);  // SEQ x DIM
    unsigned short* ybf  = (unsigned short*)(ws + 32 * MB);
    float*          t0   = (float*)         (ws + 34 * MB);  // SEQ x DFF (16 MB)
    float*          t1   = (float*)         (ws + 50 * MB);  // SEQ x DFF (16 MB)
    unsigned short* hbf  = (unsigned short*)(ws + 66 * MB);  // SEQ x DFF bf16 (8 MB)
    unsigned short* xbf  = (unsigned short*)(ws + 74 * MB);

    auto gemm = [&](const unsigned short* A, const float* W, float* C,
                    int M, int N, int K, const float* alpha) {
        dim3 grid(N / 128, M / 128);
        k_gemm<<<grid, 256, 0, stream>>>(A, W, C, M, N, K, alpha);
    };

    k_embed_rms<<<SEQ, 256, 0, stream>>>(ids, embed, x, ex);

    for (int l = 0; l < NL; ++l) {
        // ---- attention ----
        k_rms_bf16<<<SEQ, 256, 0, stream>>>(x, abf);
        gemm(abf, Wq + (size_t)l * DIM * DIM, qf, SEQ, DIM, DIM, nullptr);
        gemm(abf, Wk + (size_t)l * DIM * DIM, kf, SEQ, DIM, DIM, nullptr);
        gemm(abf, Wv + (size_t)l * DIM * DIM, vf, SEQ, DIM, DIM, nullptr);

        dim3 pgrid(SEQ, NH / 4), pblk(64, 4);
        k_qkv_post<<<pgrid, pblk, 0, stream>>>(qf, qhB, head_g + l * NH, 0);
        k_qkv_post<<<pgrid, pblk, 0, stream>>>(kf, khB, nullptr, 1);
        k_qkv_post<<<pgrid, pblk, 0, stream>>>(vf, vTB, nullptr, 2);

        k_attn<<<dim3(SEQ / 16, NH), 32, 0, stream>>>(qhB, khB, vTB, yF);

        k_to_bf16<<<(SEQ * DIM) / 256, 256, 0, stream>>>(yF, ybf, SEQ * DIM);
        gemm(ybf, Wo + (size_t)l * DIM * DIM, t0, SEQ, DIM, DIM, nullptr);
        k_add_scaled<<<(SEQ * DIM) / 256, 256, 0, stream>>>(x, t0, attn_g + l, SEQ * DIM);

        // ---- MLP ----
        k_rms_bf16<<<SEQ, 256, 0, stream>>>(x, abf);
        gemm(abf, Wg + (size_t)l * DFF * DIM, t0, SEQ, DFF, DIM, nullptr);
        gemm(abf, Wl + (size_t)l * DFF * DIM, t1, SEQ, DFF, DIM, nullptr);
        k_silu_mul<<<(SEQ * DFF) / 256, 256, 0, stream>>>(t0, t1, hbf, SEQ * DFF);
        gemm(hbf, Wr + (size_t)l * DIM * DFF, t0, SEQ, DIM, DFF, nullptr);
        k_add_scaled<<<(SEQ * DIM) / 256, 256, 0, stream>>>(x, t0, mlp_g + l, SEQ * DIM);

        // ---- embed skip ----
        k_add_scaled<<<(SEQ * DIM) / 256, 256, 0, stream>>>(x, ex, emb_skip + l, SEQ * DIM);
    }

    // ---- LM head ----
    k_rms_bf16<<<SEQ, 256, 0, stream>>>(x, xbf);
    gemm(xbf, lm_head, (float*)d_out, SEQ, VOC, DIM, lm_gain);
}
